// TDFDCNNDetector_88819923681773
// MI455X (gfx1250) — matrix operand order repaired and checked by reading, not compiled, not run on hardware
//
#include <hip/hip_runtime.h>
#include <stdint.h>

// ---------------- types for WMMA (gfx1250, wave32) ----------------
typedef __bf16 v16bf __attribute__((ext_vector_type(16)));
typedef __bf16 v8bf  __attribute__((ext_vector_type(8)));
typedef float  v8f   __attribute__((ext_vector_type(8)));

#define NPOS 286720        // 20 * 14 * 1024  (S * NSYM * FFT)
#define NCOL 280           // 20 * 14
#define FFTN 1024
#define NB   280           // batch B = 20*14
#define RACT 624

// =================================================================
// 1024-point radix-2 DIT FFT in LDS. 256 threads. sign=-1 fwd, +1 inv.
// =================================================================
__device__ inline void fft1024_lds(float2* sd, int tid, float sign) {
  __syncthreads();
  int shift = 0;
  for (int len = 2; len <= FFTN; len <<= 1) {
    int half = len >> 1;
    #pragma unroll 1
    for (int b = tid; b < (FFTN / 2); b += 256) {
      int pos = b & (half - 1);
      int grp = b >> shift;
      int i0 = grp * len + pos;
      int i1 = i0 + half;
      float ang = sign * 6.283185307179586f * (float)pos / (float)len;
      float sv, cv;
      __sincosf(ang, &sv, &cv);
      float2 u = sd[i0], v = sd[i1];
      float tr = cv * v.x - sv * v.y;
      float ti = cv * v.y + sv * v.x;
      sd[i1] = make_float2(u.x - tr, u.y - ti);
      sd[i0] = make_float2(u.x + tr, u.y + ti);
    }
    ++shift;
    __syncthreads();
  }
}

// ---------------- IFFT: rx_prob (B,8,624,1) -> bufA bf16 [8][20][14][1024]
__global__ void k_ifft(const float* __restrict__ rx, __bf16* __restrict__ bufA) {
  __shared__ float2 sd[FFTN];
  int bi = blockIdx.x;          // 0..1119
  int b  = bi >> 2;             // batch 0..279
  int a  = bi & 3;              // antenna
  int tid = threadIdx.x;
  for (int i = tid; i < FFTN; i += 256) {
    int j = (int)(__brev((unsigned)i) >> 22);   // 10-bit bit reverse
    float re = 0.f, im = 0.f;
    if (i < RACT) {
      re = rx[((size_t)b * 8 + 2 * a) * RACT + i];
      im = rx[((size_t)b * 8 + 2 * a + 1) * RACT + i];
    }
    sd[j] = make_float2(re, im);
  }
  fft1024_lds(sd, tid, +1.0f);                  // inverse (conjugate twiddles)
  int s = b / 14, sym = b % 14;
  size_t baseR = ((size_t)(2 * a) * NCOL + s * 14 + sym) * FFTN;
  size_t baseI = ((size_t)(2 * a + 1) * NCOL + s * 14 + sym) * FFTN;
  const float inv = 1.0f / (float)FFTN;
  for (int t = tid; t < FFTN; t += 256) {
    float2 v = sd[t];
    bufA[baseR + t] = (__bf16)(v.x * inv);
    bufA[baseI + t] = (__bf16)(v.y * inv);
  }
}

// ---------------- forward FFT: tdout f32 [8][20][14][1024] -> rxfd (280,8,624)
__global__ void k_fft(const float* __restrict__ td, float* __restrict__ rxfd) {
  __shared__ float2 sd[FFTN];
  int bi = blockIdx.x;
  int b  = bi >> 2;
  int a  = bi & 3;
  int tid = threadIdx.x;
  int s = b / 14, sym = b % 14;
  size_t baseR = ((size_t)(2 * a) * NCOL + s * 14 + sym) * FFTN;
  size_t baseI = ((size_t)(2 * a + 1) * NCOL + s * 14 + sym) * FFTN;
  for (int i = tid; i < FFTN; i += 256) {
    int j = (int)(__brev((unsigned)i) >> 22);
    sd[j] = make_float2(td[baseR + i], td[baseI + i]);
  }
  fft1024_lds(sd, tid, -1.0f);                  // forward
  for (int r = tid; r < RACT; r += 256) {
    float2 v = sd[r];
    rxfd[((size_t)b * 8 + 2 * a) * RACT + r]     = v.x;
    rxfd[((size_t)b * 8 + 2 * a + 1) * RACT + r] = v.y;
  }
}

// =================================================================
// Weight repack: f32 OIHW [Cout][Cin][3][3] (or 1x1) -> bf16 [Cout][Kpad],
// tap-major K: k = (kh*3+kw)*Cin + cin, zero-padded to Kpad (mult of 32).
// Cin is a power of two (8/64/128/256) -> shifts & masks only.
// =================================================================
__global__ void k_repack(const float* __restrict__ src, __bf16* __restrict__ dst,
                         int Cout, int cb, int taps, int Kpad) {
  int i = blockIdx.x * blockDim.x + threadIdx.x;
  if (i >= Cout * Kpad) return;
  int cout = i / Kpad;
  int k    = i - cout * Kpad;
  int Cin  = 1 << cb;
  int Ktot = Cin * taps;
  float v = 0.f;
  if (k < Ktot) {
    int tap = k >> cb;
    int cin = k & (Cin - 1);
    v = src[((size_t)cout * Cin + cin) * taps + tap];
  }
  dst[i] = (__bf16)v;
}

// =================================================================
// Implicit-GEMM conv via WMMA bf16 -> f32.
// act: bf16 planes [Cin][S=20][W=14][H=1024]; wgt: bf16 [Cout][Kpad],
// tap-major K. Wave tile: 16 cout x 32 h (two 16x16 C tiles, shared A).
// B-chunks (32k x 32n) staged in LDS ([n][k], stride 40 -> 16B-aligned,
// conflict-free rows), fragments read as 128-bit LDS/global loads.
// =================================================================
__global__ void k_conv_wmma(const __bf16* __restrict__ act,
                            const __bf16* __restrict__ wgt,
                            float* __restrict__ out,
                            int cb, int taps, int Kpad) {
  const int Cin  = 1 << cb;
  const int Ktot = Cin * taps;
  const int tid  = threadIdx.x;
  const int lane = tid & 31;
  const int wave = tid >> 5;
  const int WPB  = blockDim.x >> 5;

  const int p    = blockIdx.x;          // spatial tile id
  const int col  = p >> 5;              // s*14 + w
  const int h0   = (p & 31) << 5;       // 32-wide h tile
  const int scol = col / 14;
  const int wcol = col - scol * 14;
  const int coutBase = (blockIdx.y * WPB + wave) * 16;
  const int pad = (taps == 9) ? 1 : 0;

  __shared__ __bf16 ldsB[32 * 40];      // [n=0..31][kk stride 40]

  v8f acc0 = {}, acc1 = {};
  for (int k0 = 0; k0 < Ktot; k0 += 32) {
    // ---- stage B chunk (32 k x 32 n) into LDS ----
    for (int e = tid; e < 1024; e += blockDim.x) {
      int kk = e >> 5, n = e & 31;
      int k = k0 + kk;
      __bf16 v = (__bf16)0.0f;
      if (k < Ktot) {
        int tap = k >> cb;              // 0..8 (0 for 1x1)
        int cin = k & (Cin - 1);
        int kh  = (tap * 11) >> 5;      // tap/3 for tap in 0..8
        int kw  = tap - kh * 3;
        int h = h0 + n + kh - pad;
        int w = wcol + kw - pad;
        if (h >= 0 && h < FFTN && w >= 0 && w < 14)
          v = act[((size_t)cin * NCOL + scol * 14 + w) * FFTN + h];
      }
      ldsB[n * 40 + kk] = v;
    }
    __syncthreads();

    // ---- A fragment: two aligned 16B runs (K {0..7|8..15}, {16..23|24..31}) ----
    const __bf16* wrow =
        wgt + (size_t)(coutBase + (lane & 15)) * Kpad + k0 + 8 * (lane >> 4);
    v8bf wa = *(const v8bf*)wrow;
    v8bf wb = *(const v8bf*)(wrow + 16);
    v16bf afrag = __builtin_shufflevector(wa, wb, 0, 1, 2, 3, 4, 5, 6, 7,
                                          8, 9, 10, 11, 12, 13, 14, 15);

    // ---- B fragments: two 16B runs per lane-half (K {0..7|8..15}, {16..23|24..31}), 16B-aligned LDS ----
    const __bf16* br0 = ldsB + (lane & 15) * 40 + 8 * (lane >> 4);
    v8bf p0 = *(const v8bf*)br0;
    v8bf p1 = *(const v8bf*)(br0 + 16);
    v16bf bf0 = __builtin_shufflevector(p0, p1, 0, 1, 2, 3, 4, 5, 6, 7,
                                        8, 9, 10, 11, 12, 13, 14, 15);
    const __bf16* br1 = br0 + 16 * 40;
    v8bf p2 = *(const v8bf*)br1;
    v8bf p3 = *(const v8bf*)(br1 + 16);
    v16bf bf1 = __builtin_shufflevector(p2, p3, 0, 1, 2, 3, 4, 5, 6, 7,
                                        8, 9, 10, 11, 12, 13, 14, 15);
    __syncthreads();

    acc0 = __builtin_amdgcn_wmma_f32_16x16x32_bf16(false, afrag, false, bf0,
                                                   (short)0, acc0, false, false);
    acc1 = __builtin_amdgcn_wmma_f32_16x16x32_bf16(false, afrag, false, bf1,
                                                   (short)0, acc1, false, false);
  }

  // ---- store C: VGPR v -> M = v + 8*(lane/16), N = lane%16 ----
  const int mofs = (lane >> 4) * 8;
  const int n    = lane & 15;
  #pragma unroll
  for (int v = 0; v < 8; ++v) {
    int cout = coutBase + mofs + v;
    float* op = out + ((size_t)cout * NCOL + col) * FFTN + h0;
    op[n]      = acc0[v];
    op[16 + n] = acc1[v];
  }
}

// ---------------- per-channel batch stats -> (scale, shift); deterministic
__global__ void k_bnstats(const float* __restrict__ conv,
                          const float* __restrict__ g,
                          const float* __restrict__ be,
                          float* __restrict__ bnp) {
  int c = blockIdx.x;
  const float* p = conv + (size_t)c * NPOS;
  float s = 0.f, q = 0.f;
  for (int i = threadIdx.x; i < NPOS; i += 256) {
    float x = p[i];
    s += x; q += x * x;
  }
  __shared__ float ss[256], sq[256];
  ss[threadIdx.x] = s; sq[threadIdx.x] = q;
  __syncthreads();
  for (int st = 128; st > 0; st >>= 1) {
    if (threadIdx.x < st) {
      ss[threadIdx.x] += ss[threadIdx.x + st];
      sq[threadIdx.x] += sq[threadIdx.x + st];
    }
    __syncthreads();
  }
  if (threadIdx.x == 0) {
    float mean = ss[0] / (float)NPOS;
    float var  = sq[0] / (float)NPOS - mean * mean;
    float sc   = g[c] * rsqrtf(var + 1e-5f);
    bnp[2 * c]     = sc;
    bnp[2 * c + 1] = be[c] - mean * sc;
  }
}

// ---------------- BN apply + ReLU -> bf16
__global__ void k_bn_relu(const float* __restrict__ conv,
                          const float* __restrict__ bnp,
                          __bf16* __restrict__ dst, size_t tot) {
  size_t i = (size_t)blockIdx.x * blockDim.x + threadIdx.x;
  if (i >= tot) return;
  int c = (int)(i / NPOS);
  float y = bnp[2 * c] * conv[i] + bnp[2 * c + 1];
  dst[i] = (__bf16)fmaxf(y, 0.f);
}

// ---------------- BN apply + shortcut add + ReLU -> bf16
__global__ void k_bn_add_relu(const float* __restrict__ conv,
                              const float* __restrict__ idt,
                              const float* __restrict__ bnp,
                              __bf16* __restrict__ dst, size_t tot) {
  size_t i = (size_t)blockIdx.x * blockDim.x + threadIdx.x;
  if (i >= tot) return;
  int c = (int)(i / NPOS);
  float y = bnp[2 * c] * conv[i] + bnp[2 * c + 1] + idt[i];
  dst[i] = (__bf16)fmaxf(y, 0.f);
}

// ---------------- 1x1 out-conv 256->8 + bias, f32 out
__global__ void k_outconv(const __bf16* __restrict__ act,
                          const float* __restrict__ w,     // [8][256]
                          const float* __restrict__ bias,
                          float* __restrict__ td) {
  size_t pos = (size_t)blockIdx.x * blockDim.x + threadIdx.x;
  if (pos >= NPOS) return;
  float acc[8];
  #pragma unroll
  for (int o = 0; o < 8; ++o) acc[o] = bias[o];
  for (int c = 0; c < 256; ++c) {
    float x = (float)act[(size_t)c * NPOS + pos];
    #pragma unroll
    for (int o = 0; o < 8; ++o) acc[o] += w[o * 256 + c] * x;
  }
  #pragma unroll
  for (int o = 0; o < 8; ++o) td[(size_t)o * NPOS + pos] = acc[o];
}

// ---------------- head: (K=5,1) convs along R
__global__ void k_fc1(const float* __restrict__ rx, const float* __restrict__ w,
                      const float* __restrict__ bias, float* __restrict__ h1) {
  size_t i = (size_t)blockIdx.x * blockDim.x + threadIdx.x;
  if (i >= (size_t)NB * 64 * RACT) return;
  int r = (int)(i % RACT);
  size_t t1 = i / RACT;
  int co = (int)(t1 % 64);
  int b  = (int)(t1 / 64);
  float acc = bias[co];
  for (int ci = 0; ci < 8; ++ci) {
    const float* xp = rx + ((size_t)b * 8 + ci) * RACT;
    const float* wp = w + ((size_t)co * 8 + ci) * 5;
    #pragma unroll
    for (int t = 0; t < 5; ++t) {
      int rr = r + t - 2;
      if (rr >= 0 && rr < RACT) acc += wp[t] * xp[rr];
    }
  }
  h1[i] = fmaxf(acc, 0.f);
}

__global__ void k_fc2(const float* __restrict__ h1, const float* __restrict__ w,
                      const float* __restrict__ bias, float* __restrict__ h2) {
  size_t i = (size_t)blockIdx.x * blockDim.x + threadIdx.x;
  if (i >= (size_t)NB * 64 * RACT) return;
  int r = (int)(i % RACT);
  size_t t1 = i / RACT;
  int co = (int)(t1 % 64);
  int b  = (int)(t1 / 64);
  float acc = bias[co];
  for (int ci = 0; ci < 64; ++ci) {
    const float* xp = h1 + ((size_t)b * 64 + ci) * RACT;
    const float* wp = w + ((size_t)co * 64 + ci) * 5;
    #pragma unroll
    for (int t = 0; t < 5; ++t) {
      int rr = r + t - 2;
      if (rr >= 0 && rr < RACT) acc += wp[t] * xp[rr];
    }
  }
  h2[i] = fmaxf(acc, 0.f);
}

__global__ void k_fc3(const float* __restrict__ h2, const float* __restrict__ w,
                      const float* __restrict__ bias, float* __restrict__ outp) {
  size_t i = (size_t)blockIdx.x * blockDim.x + threadIdx.x;
  if (i >= (size_t)NB * 4 * RACT) return;
  int r = (int)(i % RACT);
  size_t t1 = i / RACT;
  int co = (int)(t1 % 4);
  int b  = (int)(t1 / 4);
  float acc = bias[co];
  for (int ci = 0; ci < 64; ++ci) {
    const float* xp = h2 + ((size_t)b * 64 + ci) * RACT;
    const float* wp = w + ((size_t)co * 64 + ci) * 5;
    #pragma unroll
    for (int t = 0; t < 5; ++t) {
      int rr = r + t - 2;
      if (rr >= 0 && rr < RACT) acc += wp[t] * xp[rr];
    }
  }
  size_t oi = ((size_t)b * 4 + co) * RACT + r;
  outp[oi] = 1.0f / (1.0f + expf(-acc));                 // sigmoid(llrs)
  outp[(size_t)NB * 4 * RACT + oi] = acc;                // llrs
}

// =================================================================
extern "C" void kernel_launch(void* const* d_in, const int* in_sizes, int n_in,
                              void* d_out, int out_size, void* d_ws, size_t ws_size,
                              hipStream_t stream) {
  const float* rx_prob = (const float*)d_in[0];
  const float* out_w = (const float*)d_in[22];
  const float* out_b = (const float*)d_in[23];
  const float* fc1_w = (const float*)d_in[24];
  const float* fc1_b = (const float*)d_in[25];
  const float* fc2_w = (const float*)d_in[26];
  const float* fc2_b = (const float*)d_in[27];
  const float* fc3_w = (const float*)d_in[28];
  const float* fc3_b = (const float*)d_in[29];

  // ---- workspace carve ----
  uint8_t* ws = (uint8_t*)d_ws;
  const size_t SZ_ACT  = (size_t)256 * NPOS * sizeof(__bf16);   // 146,800,640
  const size_t SZ_F32  = (size_t)256 * NPOS * sizeof(float);    // 293,601,280
  __bf16* bufA  = (__bf16*)ws;                                  // activations x
  __bf16* bufB  = (__bf16*)(ws + SZ_ACT);                       // activations h
  float*  convf = (float*)(ws + 2 * SZ_ACT);                    // raw conv out
  float*  idtf  = (float*)(ws + 2 * SZ_ACT + SZ_F32);           // shortcut out
  __bf16* wbuf  = (__bf16*)(ws + 2 * SZ_ACT + 2 * SZ_F32);      // bf16 weights (<=1.2MB)
  float*  bnp   = (float*)(ws + 2 * SZ_ACT + 2 * SZ_F32 + (2u << 20));
  // post-conv buffers reuse convf region (convf dead after last bn apply)
  float* tdout = convf;                                          //  9.2 MB
  float* rxfd  = (float*)((uint8_t*)convf + (16u << 20));        //  5.6 MB
  float* h1    = (float*)((uint8_t*)convf + (48u << 20));        // 44.7 MB
  float* h2    = (float*)((uint8_t*)convf + (96u << 20));        // 44.7 MB

  auto conv = [&](const __bf16* x, int cb, int Cout, const float* wf,
                  float* outf, int taps) {
    int Cin  = 1 << cb;
    int Kpad = (Cin * taps + 31) & ~31;
    int nw   = Cout * Kpad;
    k_repack<<<(nw + 255) / 256, 256, 0, stream>>>(wf, wbuf, Cout, cb, taps, Kpad);
    int WPB = (Cout >= 128) ? 8 : (Cout / 16);
    dim3 grid(NCOL * 32, Cout / (16 * WPB));
    k_conv_wmma<<<grid, WPB * 32, 0, stream>>>(x, wbuf, outf, cb, taps, Kpad);
  };

  auto resblock = [&](int cb, int Cout, int base) {
    const float* w1  = (const float*)d_in[base + 0];
    const float* g1  = (const float*)d_in[base + 1];
    const float* be1 = (const float*)d_in[base + 2];
    const float* w2  = (const float*)d_in[base + 3];
    const float* g2  = (const float*)d_in[base + 4];
    const float* be2 = (const float*)d_in[base + 5];
    const float* wsc = (const float*)d_in[base + 6];
    size_t tot = (size_t)Cout * NPOS;
    unsigned gb = (unsigned)((tot + 255) / 256);
    int cbOut = (Cout == 64) ? 6 : (Cout == 128) ? 7 : 8;
    conv(bufA, cb, Cout, w1, convf, 9);
    k_bnstats<<<Cout, 256, 0, stream>>>(convf, g1, be1, bnp);
    k_bn_relu<<<gb, 256, 0, stream>>>(convf, bnp, bufB, tot);
    conv(bufA, cb, Cout, wsc, idtf, 1);            // 1x1 shortcut
    conv(bufB, cbOut, Cout, w2, convf, 9);
    k_bnstats<<<Cout, 256, 0, stream>>>(convf, g2, be2, bnp);
    k_bn_add_relu<<<gb, 256, 0, stream>>>(convf, idtf, bnp, bufA, tot);
  };

  // ---- pipeline ----
  k_ifft<<<NB * 4, 256, 0, stream>>>(rx_prob, bufA);
  resblock(3, 64, 1);      // Cin=8
  resblock(6, 128, 8);     // Cin=64
  resblock(7, 256, 15);    // Cin=128
  k_outconv<<<(NPOS + 255) / 256, 256, 0, stream>>>(bufA, out_w, out_b, tdout);
  k_fft<<<NB * 4, 256, 0, stream>>>(tdout, rxfd);

  size_t n1 = (size_t)NB * 64 * RACT;
  k_fc1<<<(unsigned)((n1 + 255) / 256), 256, 0, stream>>>(rxfd, fc1_w, fc1_b, h1);
  k_fc2<<<(unsigned)((n1 + 255) / 256), 256, 0, stream>>>(h1, fc2_w, fc2_b, h2);
  size_t n3 = (size_t)NB * 4 * RACT;
  k_fc3<<<(unsigned)((n3 + 255) / 256), 256, 0, stream>>>(h2, fc3_w, fc3_b, (float*)d_out);
}
